// GAU_4982162063611
// MI455X (gfx1250) — matrix operand order repaired and checked by reading, not compiled, not run on hardware
//
#include <hip/hip_runtime.h>

// ---------------------------------------------------------------------------
// GAU block for MI455X (gfx1250, wave32, WMMA, async global->LDS staging).
// Pipeline (f16 operands, f32 accum; ~181 MB of intermediates live in 192MB L2):
//   [cvt]  W1^T, W2^T -> f16 ; rope sin/cos tables
//   [norm] h = rmsnorm(x) -> f16
//   [G1]   uv = silu(h@W1+b1); split -> u16, vT16 (transposed), q16, k16
//   [G2]   S  = (q@k^T)/512 -> f32
//   [rope] kern = relu(rope(S))^2 * 2^30 -> f16  (2^30 avoids f16 underflow:
//          raw relu^2 values are ~1e-12, far below f16 min normal)
//   [G3]   g  = u * (kern@v) * 2^-30 -> f16   (B = vT16, K-contiguous)
//   [G4]   out = g@W2 + b2 + x -> f32         (B = w2t, K-contiguous)
// Every GEMM consumes B as [N,K] row-major so LDS staging is pure 16B copies,
// done with GLOBAL_LOAD_ASYNC_TO_LDS_B128 + s_wait_asynccnt, double-buffered.
// ---------------------------------------------------------------------------

typedef __attribute__((ext_vector_type(16))) _Float16 v16h;
typedef __attribute__((ext_vector_type(8)))  _Float16 h8;
typedef __attribute__((ext_vector_type(8)))  float    v8f;

#define DIM      512
#define LEN      512
#define SHARED_D 128
#define EXPAND   1024
#define PROJ     2176
#define NBATCH   32
#define EPS_F    1e-6f
#define KERN_SCALE      1073741824.0f          // 2^30
#define KERN_INV_SCALE  9.313225746154785e-10f // 2^-30

// async 16B global -> LDS copy (ASYNCcnt-tracked, no VGPR round trip).
// lds_off: wave-relative LDS byte address (low 32 bits of flat LDS pointer).
__device__ __forceinline__ void async_ldst16(unsigned lds_off, const void* gaddr) {
  unsigned long long ga = (unsigned long long)(uintptr_t)gaddr;
  asm volatile("global_load_async_to_lds_b128 %0, %1, off"
               :: "v"(lds_off), "v"(ga) : "memory");
}
__device__ __forceinline__ void wait_async0() {
  asm volatile("s_wait_asynccnt 0" ::: "memory");
}
__device__ __forceinline__ unsigned lds_off_u32(const void* p) {
  return (unsigned)(uintptr_t)p;   // LDS aperture: addr[31:0] = LDS offset
}

// ---------------------------------------------------------------------------
// small helper kernels
// ---------------------------------------------------------------------------
__global__ void k_cvt_tr_f16(const float* __restrict__ s, _Float16* __restrict__ d,
                             int N, int Klog2) {
  // src is [K,N] row-major f32; dst is [N,K] row-major f16
  int i = blockIdx.x * blockDim.x + threadIdx.x;   // over N*K
  int K = 1 << Klog2;
  if (i >= N * K) return;
  int n = i >> Klog2;
  int k = i & (K - 1);
  d[i] = (_Float16)s[(size_t)k * N + n];
}

__global__ void k_rope_tables(float* __restrict__ sinT, float* __restrict__ cosT) {
  int i = blockIdx.x * blockDim.x + threadIdx.x;   // LEN*256 threads
  if (i >= LEN * 256) return;
  int pos = i >> 8;
  int j   = i & 255;
  float invf = expf(-(float)j * (1.0f / 256.0f) * 9.210340371976184f); // 10000^(-j/256)
  float a = (float)pos * invf;
  sinT[i] = sinf(a);
  cosT[i] = cosf(a);
}

__global__ __launch_bounds__(256) void k_rmsnorm(const float* __restrict__ x,
                                                 const float* __restrict__ norm_scale,
                                                 _Float16* __restrict__ h16) {
  __shared__ float red[256];
  const int row = blockIdx.x;           // 0 .. NBATCH*LEN-1
  const int t   = threadIdx.x;
  const size_t base = (size_t)row * DIM;
  float x0 = x[base + t];
  float x1 = x[base + t + 256];
  red[t] = x0 * x0 + x1 * x1;
  __syncthreads();
  for (int off = 128; off > 0; off >>= 1) {
    if (t < off) red[t] += red[t + off];
    __syncthreads();
  }
  float inv = rsqrtf(red[0] * (1.0f / DIM) + EPS_F) * norm_scale[0];
  h16[base + t]       = (_Float16)(x0 * inv);
  h16[base + t + 256] = (_Float16)(x1 * inv);
}

__global__ void k_rope_relu2(const float* __restrict__ S,
                             const float* __restrict__ sinT,
                             const float* __restrict__ cosT,
                             _Float16* __restrict__ kern) {
  int i = blockIdx.x * blockDim.x + threadIdx.x;   // NBATCH*LEN*256 threads
  if (i >= NBATCH * LEN * 256) return;
  int j = i & 255;
  int n = (i >> 8) & (LEN - 1);
  int b = i >> 17;
  size_t rbase = ((size_t)b * LEN + n) * LEN;
  float x1 = S[rbase + j];
  float x2 = S[rbase + j + 256];
  float ct = cosT[n * 256 + j];
  float st = sinT[n * 256 + j];
  float r1 = x1 * ct - x2 * st;
  float r2 = x2 * ct + x1 * st;
  r1 = fmaxf(r1, 0.0f);
  r2 = fmaxf(r2, 0.0f);
  kern[rbase + j]       = (_Float16)(r1 * r1 * KERN_SCALE);
  kern[rbase + j + 256] = (_Float16)(r2 * r2 * KERN_SCALE);
}

// ---------------------------------------------------------------------------
// Tiled WMMA GEMM: C[M,N] = A[M,K] (f16 row-major) x B[N,K] (f16 row-major,
// i.e. K-contiguous / pre-transposed), f32 accum.
// Block tile 128x128, K-step 32; 256 threads = 8 waves (4M x 2N),
// wave tile 32x64 = 2x4 fragments of 16x16.
// LDS double-buffered; tiles staged with global_load_async_to_lds_b128 and
// fenced with s_wait_asynccnt before the workgroup barrier, so tile kt+1
// streams in while the WMMAs for tile kt run.
// Fragment layouts per CDNA5 ISA:
//   A lane l: row = base+(l&15); halves = K[(l>>4)*8 + 0..7], K[(l>>4)*8+16..23]
//   B lane l: col = base+(l&15); halves = K[(l>>4)*8 + 0..7], K[(l>>4)*8+16..23]
//   C lane l, vgpr r: M = r + 8*(l>>4), N = (l&15)
// MODE: 1 = uv-split+silu  2 = qk score  3 = kern@v (*u, descale)  4 = final
// ---------------------------------------------------------------------------
struct GemmArgs {
  const _Float16* A;
  const _Float16* B;      // [N,K] row-major
  size_t strideA;         // per-batch element stride (0 if shared)
  size_t strideB;
  int K;
  int N;
  const float* bias;      // b1 (mode1) / b2 (mode4)
  const float* gamma;     // mode1: [2,128]
  const float* beta;      // mode1: [2,128]
  const float* resid;     // mode4: x (flat)
  const _Float16* u;      // mode3: u16
  _Float16* oU;           // mode1: u16
  _Float16* oVt;          // mode1: v transposed [b, e, n]
  _Float16* oQ;           // mode1: q16
  _Float16* oK;           // mode1: k16
  _Float16* oH;           // mode3: g16
  float*    oF;           // mode2: S / mode4: out
};

#define LDS_STRIDE 40     // 32 + 8 halves padding vs bank conflicts

template <int MODE>
__global__ __launch_bounds__(256) void gemm_wmma(GemmArgs args) {
  __shared__ _Float16 As[2][128 * LDS_STRIDE];
  __shared__ _Float16 Bs[2][128 * LDS_STRIDE];

  const int tid  = threadIdx.x;
  const int z    = blockIdx.z;
  const int K    = args.K;
  const _Float16* __restrict__ Ab = args.A + (size_t)z * args.strideA;
  const _Float16* __restrict__ Bb = args.B + (size_t)z * args.strideB;

  const int wave = tid >> 5;
  const int lane = tid & 31;
  const int lrow = lane & 15;
  const int lhi  = lane >> 4;
  const int wm   = wave & 3;   // wave M offset = wm*32
  const int wn   = wave >> 2;  // wave N offset = wn*64

  // staging coordinates: each thread moves 16 halves of one A row and one B row
  const int trow = tid >> 1;            // 0..127
  const int toff = (tid & 1) * 16;      // 0 / 16 halves
  const size_t aRowBase = (size_t)(blockIdx.y * 128 + trow) * K + toff;
  const size_t bRowBase = (size_t)(blockIdx.x * 128 + trow) * K + toff;
  const unsigned ldsA = lds_off_u32(&As[0][trow * LDS_STRIDE + toff]);
  const unsigned ldsB = lds_off_u32(&Bs[0][trow * LDS_STRIDE + toff]);
  const unsigned bufBytes = (unsigned)(128 * LDS_STRIDE * sizeof(_Float16));

  v8f acc[2][4];
  const v8f zacc = {0.f, 0.f, 0.f, 0.f, 0.f, 0.f, 0.f, 0.f};
#pragma unroll
  for (int i = 0; i < 2; ++i)
#pragma unroll
    for (int j = 0; j < 4; ++j) acc[i][j] = zacc;

  const int nk = K >> 5;

  // stage tile 0 into buffer 0
  {
    async_ldst16(ldsA,      Ab + aRowBase);
    async_ldst16(ldsA + 16, Ab + aRowBase + 8);
    async_ldst16(ldsB,      Bb + bRowBase);
    async_ldst16(ldsB + 16, Bb + bRowBase + 8);
  }
  wait_async0();
  __syncthreads();

  for (int kt = 0; kt < nk; ++kt) {
    const int cur = kt & 1;
    // stage next tile into the other buffer while we compute on this one
    if (kt + 1 < nk) {
      const unsigned nb = (unsigned)((kt + 1) & 1) * bufBytes;
      const size_t ko = (size_t)(kt + 1) * 32;
      async_ldst16(ldsA + nb,      Ab + aRowBase + ko);
      async_ldst16(ldsA + nb + 16, Ab + aRowBase + ko + 8);
      async_ldst16(ldsB + nb,      Bb + bRowBase + ko);
      async_ldst16(ldsB + nb + 16, Bb + bRowBase + ko + 8);
    }

    // ---- fragments from current buffer ----
    v16h afrag[2];
    v16h bfrag[4];
#pragma unroll
    for (int f = 0; f < 2; ++f) {
      const int r = wm * 32 + f * 16 + lrow;
      const h8* p = reinterpret_cast<const h8*>(&As[cur][r * LDS_STRIDE + lhi * 8]);
      afrag[f] = __builtin_shufflevector(p[0], p[2],
          0, 1, 2, 3, 4, 5, 6, 7, 8, 9, 10, 11, 12, 13, 14, 15);
    }
#pragma unroll
    for (int f = 0; f < 4; ++f) {
      const int c = wn * 64 + f * 16 + lrow;
      const h8* p = reinterpret_cast<const h8*>(&Bs[cur][c * LDS_STRIDE + lhi * 8]);
      bfrag[f] = __builtin_shufflevector(p[0], p[2],
          0, 1, 2, 3, 4, 5, 6, 7, 8, 9, 10, 11, 12, 13, 14, 15);
    }
    // ---- 8 WMMAs ----
#pragma unroll
    for (int fm = 0; fm < 2; ++fm)
#pragma unroll
      for (int fn = 0; fn < 4; ++fn)
        acc[fm][fn] = __builtin_amdgcn_wmma_f32_16x16x32_f16(
            false, afrag[fm], false, bfrag[fn], (short)0, acc[fm][fn],
            false, false);

    wait_async0();      // next buffer's LDS writes (this wave) complete
    __syncthreads();    // everyone's complete -> safe to read next buffer
  }

  // ---- epilogue ----
#pragma unroll
  for (int fm = 0; fm < 2; ++fm) {
#pragma unroll
    for (int fn = 0; fn < 4; ++fn) {
#pragma unroll
      for (int r = 0; r < 8; ++r) {
        const int grow = blockIdx.y * 128 + wm * 32 + fm * 16 + lhi * 8 + r;
        const int gcol = blockIdx.x * 128 + wn * 64 + fn * 16 + lrow;
        float c = acc[fm][fn][r];
        if constexpr (MODE == 1) {
          c += args.bias[gcol];
          float s = c / (1.0f + __expf(-c));          // silu
          if (gcol < EXPAND) {
            args.oU[(size_t)grow * EXPAND + gcol] = (_Float16)s;
          } else if (gcol < 2 * EXPAND) {
            // store v TRANSPOSED: vT[b, e, n] so G3 sees B as [N,K] row-major
            const int b = grow >> 9;
            const int n = grow & (LEN - 1);
            const int e = gcol - EXPAND;
            args.oVt[((size_t)b * EXPAND + e) * LEN + n] = (_Float16)s;
          } else {
            const int d = gcol - 2 * EXPAND;
            args.oQ[(size_t)grow * SHARED_D + d] =
                (_Float16)(s * args.gamma[d] + args.beta[d]);
            args.oK[(size_t)grow * SHARED_D + d] =
                (_Float16)(s * args.gamma[SHARED_D + d] + args.beta[SHARED_D + d]);
          }
        } else if constexpr (MODE == 2) {
          args.oF[((size_t)z * LEN + grow) * LEN + gcol] = c * (1.0f / (float)LEN);
        } else if constexpr (MODE == 3) {
          const size_t idx = ((size_t)z * LEN + grow) * EXPAND + gcol;
          float uu = (float)args.u[idx];
          args.oH[idx] = (_Float16)(c * KERN_INV_SCALE * uu);
        } else {  // MODE == 4
          const size_t idx = (size_t)grow * DIM + gcol;
          args.oF[idx] = c + args.bias[gcol] + args.resid[idx];
        }
      }
    }
  }
}

// ---------------------------------------------------------------------------
// host orchestration
// ---------------------------------------------------------------------------
extern "C" void kernel_launch(void* const* d_in, const int* in_sizes, int n_in,
                              void* d_out, int out_size, void* d_ws, size_t ws_size,
                              hipStream_t stream) {
  (void)in_sizes; (void)n_in; (void)out_size; (void)ws_size;

  const float* x      = (const float*)d_in[0];   // [32,512,512]
  const float* W1     = (const float*)d_in[1];   // [512,2176]
  const float* b1     = (const float*)d_in[2];   // [2176]
  const float* W2     = (const float*)d_in[3];   // [1024,512]
  const float* b2     = (const float*)d_in[4];   // [512]
  const float* gamma  = (const float*)d_in[5];   // [2,128]
  const float* beta   = (const float*)d_in[6];   // [2,128]
  const float* nscale = (const float*)d_in[7];   // scalar
  float* out = (float*)d_out;                    // [32,512,512]

  // workspace carve-up (~181 MB; every chunk size is a multiple of 256B)
  char* ws = (char*)d_ws;
  size_t off = 0;
  auto take = [&](size_t bytes) { char* p = ws + off; off += bytes; return p; };
  _Float16* w1t    = (_Float16*)take((size_t)PROJ * DIM * 2);             //  2.2 MB  [2176,512]
  _Float16* w2t    = (_Float16*)take((size_t)DIM * EXPAND * 2);           //  1.0 MB  [512,1024]
  _Float16* h16    = (_Float16*)take((size_t)NBATCH * LEN * DIM * 2);     // 16.8 MB
  _Float16* u16    = (_Float16*)take((size_t)NBATCH * LEN * EXPAND * 2);  // 33.6 MB
  _Float16* vT16   = (_Float16*)take((size_t)NBATCH * EXPAND * LEN * 2);  // 33.6 MB  [b,e,n]
  _Float16* q16    = (_Float16*)take((size_t)NBATCH * LEN * SHARED_D * 2);//  4.2 MB
  _Float16* k16    = (_Float16*)take((size_t)NBATCH * LEN * SHARED_D * 2);//  4.2 MB
  float*    Sbuf   = (float*)   take((size_t)NBATCH * LEN * LEN * 4);     // 33.6 MB
  _Float16* kern16 = (_Float16*)take((size_t)NBATCH * LEN * LEN * 2);     // 16.8 MB
  _Float16* g16    = (_Float16*)take((size_t)NBATCH * LEN * EXPAND * 2);  // 33.6 MB
  float*    sinT   = (float*)   take((size_t)LEN * 256 * 4);              //  0.5 MB
  float*    cosT   = (float*)   take((size_t)LEN * 256 * 4);              //  0.5 MB

  // prep: transposed f16 weights, rope tables, rmsnorm
  k_cvt_tr_f16<<<(PROJ * DIM + 255) / 256, 256, 0, stream>>>(W1, w1t, PROJ, 9);
  k_cvt_tr_f16<<<(DIM * EXPAND + 255) / 256, 256, 0, stream>>>(W2, w2t, DIM, 10);
  k_rope_tables<<<(LEN * 256) / 256, 256, 0, stream>>>(sinT, cosT);
  k_rmsnorm<<<NBATCH * LEN, 256, 0, stream>>>(x, nscale, h16);

  // G1: uv = silu(h @ W1 + b1) -> u, vT, q, k   M=16384 K=512 N=2176
  {
    GemmArgs a{};
    a.A = h16; a.B = w1t; a.strideA = 0; a.strideB = 0;
    a.K = DIM; a.N = PROJ;
    a.bias = b1; a.gamma = gamma; a.beta = beta;
    a.oU = u16; a.oVt = vT16; a.oQ = q16; a.oK = k16;
    gemm_wmma<1><<<dim3(PROJ / 128, (NBATCH * LEN) / 128, 1), 256, 0, stream>>>(a);
  }
  // G2: S = q @ k^T / 512                       per-batch M=N=512 K=128
  {
    GemmArgs a{};
    a.A = q16; a.B = k16;
    a.strideA = (size_t)LEN * SHARED_D; a.strideB = (size_t)LEN * SHARED_D;
    a.K = SHARED_D; a.N = LEN;
    a.oF = Sbuf;
    gemm_wmma<2><<<dim3(LEN / 128, LEN / 128, NBATCH), 256, 0, stream>>>(a);
  }
  // RoPE + relu^2 (+2^30 scaling) -> kern16
  k_rope_relu2<<<(NBATCH * LEN * 256) / 256, 256, 0, stream>>>(Sbuf, sinT, cosT, kern16);

  // G3: g = u * (kern @ v) * 2^-30              per-batch M=512 K=512 N=1024
  {
    GemmArgs a{};
    a.A = kern16; a.B = vT16;
    a.strideA = (size_t)LEN * LEN; a.strideB = (size_t)EXPAND * LEN;
    a.K = LEN; a.N = EXPAND;
    a.u = u16; a.oH = g16;
    gemm_wmma<3><<<dim3(EXPAND / 128, LEN / 128, NBATCH), 256, 0, stream>>>(a);
  }
  // G4: out = g @ W2 + b2 + x                   M=16384 K=1024 N=512
  {
    GemmArgs a{};
    a.A = g16; a.B = w2t; a.strideA = 0; a.strideB = 0;
    a.K = EXPAND; a.N = DIM;
    a.bias = b2; a.resid = x; a.oF = out;
    gemm_wmma<4><<<dim3(DIM / 128, (NBATCH * LEN) / 128, 1), 256, 0, stream>>>(a);
  }
}
